// BitNetTransformer_19310172963451
// MI455X (gfx1250) — matrix operand order repaired and checked by reading, not compiled, not run on hardware
//
#include <hip/hip_runtime.h>
#include <hip/hip_bf16.h>

// BitNet transformer forward for MI455X (gfx1250, wave32, WMMA).
// Matmuls via v_wmma_f32_16x16x32_f16; weight tiles staged to LDS with
// CDNA5 async global->LDS DMA (global_load_async_to_lds_b128 / s_wait_asynccnt).

typedef __attribute__((ext_vector_type(16))) _Float16 v16h;
typedef __attribute__((ext_vector_type(8)))  float    v8f;

#define DIMX 1024
#define NSEQ 1024
#define QBF  128.0f
#define EPSF 1e-5f

__device__ __forceinline__ v8f wmma16(v16h a, v16h b, v8f c) {
  return __builtin_amdgcn_wmma_f32_16x16x32_f16(false, a, false, b, (short)0, c, false, false);
}

// A-fragment (16x32 f16, M x K), row-major src with leading dim ld (f16 elems).
__device__ __forceinline__ v16h load_A(const _Float16* __restrict__ src, int ld) {
  const int lane = threadIdx.x & 31;
  const _Float16* p = src + (size_t)(lane & 15) * ld + ((lane >> 4) << 3);
  union { v16h h; unsigned int u[8]; } r;
#pragma unroll
  for (int v = 0; v < 8; ++v) {
    const int k = ((v & 4) << 2) + ((v & 3) << 1);   // 16*(v>=4) + 2*(v%4)
    r.u[v] = *reinterpret_cast<const unsigned int*>(p + k);
  }
  return r.h;
}

// B-fragment (32x16 f16, K x N) where B[k][n] = src[n*ld + k].
__device__ __forceinline__ v16h load_B_T(const _Float16* __restrict__ src, int ld) {
  const int lane = threadIdx.x & 31;
  const _Float16* p = src + (size_t)(lane & 15) * ld + ((lane >> 4) << 3);
  union { v16h h; uint4 q[2]; } r;
  r.q[0] = *reinterpret_cast<const uint4*>(p);
  r.q[1] = *reinterpret_cast<const uint4*>(p + 16);
  return r.h;
}

// CDNA5 async DMA: global -> LDS, 16B per lane, tracked by ASYNCcnt.
__device__ __forceinline__ void async_g2l_b128(_Float16* lds_dst, const _Float16* gsrc) {
  unsigned int l = (unsigned int)(size_t)lds_dst;                  // LDS byte address = addr[31:0]
  unsigned long long g = (unsigned long long)(size_t)gsrc;
  asm volatile("global_load_async_to_lds_b128 %0, %1, off" :: "v"(l), "v"(g) : "memory");
}
__device__ __forceinline__ void wait_async0() {
  asm volatile("s_wait_asynccnt 0x0" ::: "memory");
}

// ---------------- weight prep: alpha=mean(w), beta=mean|w|, sign binarize ----------------
__global__ __launch_bounds__(256) void k_wpart(const float* __restrict__ w, float* __restrict__ part) {
  __shared__ float rs[256], ra[256];
  const int t = threadIdx.x;
  const int base = blockIdx.x * 4096;
  float s = 0.f, a = 0.f;
#pragma unroll
  for (int i = 0; i < 16; ++i) { float v = w[base + i * 256 + t]; s += v; a += fabsf(v); }
  rs[t] = s; ra[t] = a; __syncthreads();
  for (int o = 128; o > 0; o >>= 1) { if (t < o) { rs[t] += rs[t + o]; ra[t] += ra[t + o]; } __syncthreads(); }
  if (t == 0) { part[2 * blockIdx.x] = rs[0]; part[2 * blockIdx.x + 1] = ra[0]; }
}

__global__ __launch_bounds__(256) void k_wfinal(const float* __restrict__ part, float* __restrict__ ab) {
  __shared__ float rs[256], ra[256];
  const int t = threadIdx.x;
  rs[t] = part[2 * t]; ra[t] = part[2 * t + 1]; __syncthreads();
  for (int o = 128; o > 0; o >>= 1) { if (t < o) { rs[t] += rs[t + o]; ra[t] += ra[t + o]; } __syncthreads(); }
  if (t == 0) { ab[0] = rs[0] * (1.0f / 1048576.0f); ab[1] = ra[0] * (1.0f / 1048576.0f); }
}

__global__ __launch_bounds__(256) void k_binarize(const float* __restrict__ w, const float* __restrict__ ab,
                                                  _Float16* __restrict__ wb) {
  const int i = blockIdx.x * 256 + threadIdx.x;
  const float d = w[i] - ab[0];
  wb[i] = (_Float16)((d > 0.f) ? 1.0f : ((d < 0.f) ? -1.0f : 0.0f));
}

// ---------------- pack x -> per-head f16 [bh][n][64] and transposed [bh][64][n] ----------------
__global__ __launch_bounds__(256) void k_pack(const float* __restrict__ x, _Float16* __restrict__ xh,
                                              _Float16* __restrict__ xht) {
  const int t = blockIdx.x * 256 + threadIdx.x;        // 0..2M-1
  const int d = t & 63, n = (t >> 6) & 1023, h = (t >> 16) & 15, b = t >> 20;
  const float v = x[((size_t)(b * 1024 + n)) * 1024 + h * 64 + d];
  xh[t] = (_Float16)v;
  xht[(((size_t)(b * 16 + h) * 64) + d) * 1024 + n] = (_Float16)v;
}

// ---------------- flash attention (q=k=v), one-pass online softmax, +residual ----------------
__global__ __launch_bounds__(256) void k_flash(const _Float16* __restrict__ xh, const _Float16* __restrict__ xht,
                                               const float* __restrict__ xin, float* __restrict__ xout) {
  const int bh = blockIdx.y;
  const int wave = threadIdx.x >> 5;
  const int lane = threadIdx.x & 31;
  const int q0 = blockIdx.x * 128 + wave * 16;
  const _Float16* base = xh + (size_t)bh * 65536;   // [n][64]
  const _Float16* vt   = xht + (size_t)bh * 65536;  // [64][n]

  const v16h bq0 = load_B_T(base + q0 * 64, 64);
  const v16h bq1 = load_B_T(base + q0 * 64 + 32, 64);

  const v8f vzero = {0.f, 0.f, 0.f, 0.f, 0.f, 0.f, 0.f, 0.f};
  v8f o0 = vzero, o1 = vzero, o2 = vzero, o3 = vzero;
  float mrun = -3.0e38f, lrun = 0.f;
  const int hi8 = (lane >> 4) << 3;

  for (int j0 = 0; j0 < NSEQ; j0 += 32) {
    v8f s0 = vzero, s1 = vzero;
    s0 = wmma16(load_A(base + j0 * 64, 64), bq0, s0);
    s0 = wmma16(load_A(base + j0 * 64 + 32, 64), bq1, s0);
    s1 = wmma16(load_A(base + (j0 + 16) * 64, 64), bq0, s1);
    s1 = wmma16(load_A(base + (j0 + 16) * 64 + 32, 64), bq1, s1);
    float tmax = -3.0e38f;
#pragma unroll
    for (int v = 0; v < 8; ++v) {
      s0[v] *= 0.125f; s1[v] *= 0.125f;               // dh^-0.5
      tmax = fmaxf(tmax, fmaxf(s0[v], s1[v]));
    }
    tmax = fmaxf(tmax, __shfl_xor(tmax, 16));
    const float mnew = fmaxf(mrun, tmax);
    const float scale = __expf(mrun - mnew);
    float psum = 0.f;
    v8f p0, p1;
#pragma unroll
    for (int v = 0; v < 8; ++v) {
      p0[v] = __expf(s0[v] - mnew);
      p1[v] = __expf(s1[v] - mnew);
      psum += p0[v] + p1[v];
    }
    psum += __shfl_xor(psum, 16);
    lrun = lrun * scale + psum;
    mrun = mnew;
#pragma unroll
    for (int v = 0; v < 8; ++v) {
      const float sc = __shfl(scale, v + hi8);
      o0[v] *= sc; o1[v] *= sc; o2[v] *= sc; o3[v] *= sc;
    }
    v16h pa;
#pragma unroll
    for (int j = 0; j < 8; ++j) { pa[j] = (_Float16)p0[j]; pa[8 + j] = (_Float16)p1[j]; }
    o0 = wmma16(pa, load_B_T(vt + 0     + j0, 1024), o0);
    o1 = wmma16(pa, load_B_T(vt + 16384 + j0, 1024), o1);
    o2 = wmma16(pa, load_B_T(vt + 32768 + j0, 1024), o2);
    o3 = wmma16(pa, load_B_T(vt + 49152 + j0, 1024), o3);
  }
  const int b = bh >> 4, h = bh & 15;
  const int n16 = lane & 15;
#pragma unroll
  for (int v = 0; v < 8; ++v) {
    const float li = 1.0f / __shfl(lrun, v + hi8);
    const int q = q0 + v + hi8;
    const size_t ro = ((size_t)(b * 1024 + q)) * 1024 + h * 64;
    xout[ro + 0  + n16] = o0[v] * li + xin[ro + 0  + n16];
    xout[ro + 16 + n16] = o1[v] * li + xin[ro + 16 + n16];
    xout[ro + 32 + n16] = o2[v] * li + xin[ro + 32 + n16];
    xout[ro + 48 + n16] = o3[v] * li + xin[ro + 48 + n16];
  }
}

// ---------------- layernorm (per row of 1024), optional row absmax ----------------
__global__ __launch_bounds__(256) void k_ln(const float* __restrict__ xin, const float* __restrict__ g,
                                            const float* __restrict__ bb, float* __restrict__ out,
                                            float* __restrict__ rowamax) {
  __shared__ float red[256];
  const int r = blockIdx.x, t = threadIdx.x;
  const float* xr = xin + (size_t)r * DIMX;
  const float v0 = xr[t], v1 = xr[t + 256], v2 = xr[t + 512], v3 = xr[t + 768];
  red[t] = v0 + v1 + v2 + v3; __syncthreads();
  for (int o = 128; o > 0; o >>= 1) { if (t < o) red[t] += red[t + o]; __syncthreads(); }
  const float mean = red[0] * (1.0f / DIMX);
  __syncthreads();
  const float d0 = v0 - mean, d1 = v1 - mean, d2 = v2 - mean, d3 = v3 - mean;
  red[t] = d0 * d0 + d1 * d1 + d2 * d2 + d3 * d3; __syncthreads();
  for (int o = 128; o > 0; o >>= 1) { if (t < o) red[t] += red[t + o]; __syncthreads(); }
  const float rstd = rsqrtf(red[0] * (1.0f / DIMX) + EPSF);
  __syncthreads();
  const float y0 = d0 * rstd * g[t]       + bb[t];
  const float y1 = d1 * rstd * g[t + 256] + bb[t + 256];
  const float y2 = d2 * rstd * g[t + 512] + bb[t + 512];
  const float y3 = d3 * rstd * g[t + 768] + bb[t + 768];
  float* orow = out + (size_t)r * DIMX;
  orow[t] = y0; orow[t + 256] = y1; orow[t + 512] = y2; orow[t + 768] = y3;
  if (rowamax) {
    red[t] = fmaxf(fmaxf(fabsf(y0), fabsf(y1)), fmaxf(fabsf(y2), fabsf(y3))); __syncthreads();
    for (int o = 128; o > 0; o >>= 1) { if (t < o) red[t] = fmaxf(red[t], red[t + o]); __syncthreads(); }
    if (t == 0) rowamax[r] = red[0];
  }
}

__global__ __launch_bounds__(256) void k_rmax(const float* __restrict__ in, int n, float* __restrict__ out) {
  __shared__ float red[256];
  float m = 0.f;
  for (int i = threadIdx.x; i < n; i += 256) m = fmaxf(m, fabsf(in[i]));
  red[threadIdx.x] = m; __syncthreads();
  for (int o = 128; o > 0; o >>= 1) { if (threadIdx.x < o) red[threadIdx.x] = fmaxf(red[threadIdx.x], red[threadIdx.x + o]); __syncthreads(); }
  if (threadIdx.x == 0) out[0] = red[0];
}

__global__ __launch_bounds__(256) void k_quant(const float* __restrict__ h, const float* __restrict__ gam,
                                               _Float16* __restrict__ hq) {
  const int i = blockIdx.x * 256 + threadIdx.x;
  float v = h[i] * (QBF / gam[0]);
  v = fminf(fmaxf(v, -QBF + EPSF), QBF - EPSF);
  hq[i] = (_Float16)v;
}

// ---------------- BitLinear GEMM: y = x_q * W_b^T, LDS-staged weights (async DMA),
// fused scale/gelu/residual/absmax ----------------
#define WLD 40   // padded LDS row stride (f16) to avoid 4-way bank conflicts (80B)

__global__ __launch_bounds__(256) void k_gemm(const _Float16* __restrict__ aq, const _Float16* __restrict__ wb,
                                              const float* __restrict__ beta, const float* __restrict__ gamma,
                                              const float* __restrict__ resid, float* __restrict__ out,
                                              float* __restrict__ amaxpart, int dogelu) {
  __shared__ _Float16 wtile[2][128 * WLD];   // 2 x 10KB double-buffered weight tile (128 cols x 32 k)
  __shared__ float red[256];
  const int tid = threadIdx.x;
  const int wave = tid >> 5, lane = tid & 31;
  const int row0 = blockIdx.x * 128 + wave * 16;
  const int col0 = blockIdx.y * 128;
  const int hi8 = (lane >> 4) << 3, n16 = lane & 15;
  const int srow = tid >> 2;          // 0..63: weight row (output col) staged by this thread
  const int sch  = (tid & 3) * 8;     // 16B chunk offset in f16 elems

  // async-stage one 128x32 weight k-chunk into wtile[bf]
  auto stage = [&](int bf, int k0) {
    const _Float16* g0 = wb + (size_t)(col0 + srow) * DIMX + k0 + sch;
    const _Float16* g1 = wb + (size_t)(col0 + 64 + srow) * DIMX + k0 + sch;
    async_g2l_b128(&wtile[bf][srow * WLD + sch], g0);
    async_g2l_b128(&wtile[bf][(64 + srow) * WLD + sch], g1);
  };

  const v8f vzero = {0.f, 0.f, 0.f, 0.f, 0.f, 0.f, 0.f, 0.f};
  v8f acc[8];
#pragma unroll
  for (int c = 0; c < 8; ++c) acc[c] = vzero;

  stage(0, 0);
  wait_async0();
  __syncthreads();
  int cur = 0;
  for (int k0 = 0; k0 < DIMX; k0 += 32) {
    if (k0 + 32 < DIMX) stage(cur ^ 1, k0 + 32);          // prefetch next chunk
    const v16h a = load_A(aq + (size_t)row0 * DIMX + k0, DIMX);
    const _Float16* wt = &wtile[cur][0];
#pragma unroll
    for (int c = 0; c < 8; ++c) {
      const _Float16* wp = wt + (16 * c + n16) * WLD + hi8;
      union { v16h h; uint4 q[2]; } bu;
      bu.q[0] = *reinterpret_cast<const uint4*>(wp);
      bu.q[1] = *reinterpret_cast<const uint4*>(wp + 16);
      const v16h bf = bu.h;
      acc[c] = wmma16(a, bf, acc[c]);
    }
    wait_async0();       // own async loads landed
    __syncthreads();     // everyone's landed; safe to read next buffer / overwrite this one next iter
    cur ^= 1;
  }

  const float s = beta[0] * gamma[0] * (1.0f / QBF);
  float amax = 0.f;
#pragma unroll
  for (int c = 0; c < 8; ++c) {
#pragma unroll
    for (int v = 0; v < 8; ++v) {
      float val = acc[c][v] * s;
      if (dogelu) val = 0.5f * val * (1.0f + erff(val * 0.70710678f));
      const size_t idx = (size_t)(row0 + v + hi8) * DIMX + (col0 + 16 * c + n16);
      if (resid) val += resid[idx];
      out[idx] = val;
      amax = fmaxf(amax, fabsf(val));
    }
  }
  if (amaxpart) {
    red[tid] = amax; __syncthreads();
    for (int o = 128; o > 0; o >>= 1) { if (tid < o) red[tid] = fmaxf(red[tid], red[tid + o]); __syncthreads(); }
    if (tid == 0) amaxpart[blockIdx.y * gridDim.x + blockIdx.x] = red[0];
  }
}

extern "C" void kernel_launch(void* const* d_in, const int* in_sizes, int n_in,
                              void* d_out, int out_size, void* d_ws, size_t ws_size,
                              hipStream_t stream) {
  (void)in_sizes; (void)n_in; (void)out_size; (void)ws_size;
  const float* x0   = (const float*)d_in[0];
  const float* lng  = (const float*)d_in[1];
  const float* lnb  = (const float*)d_in[2];
  const float* w1   = (const float*)d_in[3];
  const float* w2   = (const float*)d_in[4];
  const float* fg   = (const float*)d_in[5];
  const float* fb   = (const float*)d_in[6];

  char* base = (char*)d_ws;
  size_t off = 0;
  auto take = [&](size_t bytes) -> char* {
    char* p = base + off;
    off = (off + bytes + 255) & ~(size_t)255;
    return p;
  };
  _Float16* wb    = (_Float16*)take(12ull * 1048576 * 2);   // 24 MB binarized weights
  float*    xA    = (float*)take(8388608);                  // x ping
  float*    xB    = (float*)take(8388608);                  // x pong (attn out)
  _Float16* xh    = (_Float16*)take(4194304);               // per-head f16
  _Float16* xht   = (_Float16*)take(4194304);               // per-head f16 transposed
  float*    hbuf  = (float*)take(8388608);                  // LN out / gelu out
  _Float16* hq    = (_Float16*)take(4194304);               // quantized activations
  float*    part  = (float*)take(4096);                     // reduction partials
  float*    rowam = (float*)take(8192);                     // 2048 row absmax
  float*    scal  = (float*)take(256);                      // ab[24] + gamma
  float*    gam   = scal + 32;

  // Weight prep: 12 matrices (w1[L], w2[L])
  for (int idx = 0; idx < 12; ++idx) {
    const float* w = (idx & 1) ? (w2 + (size_t)(idx >> 1) * 1048576)
                               : (w1 + (size_t)(idx >> 1) * 1048576);
    float* ab = scal + 2 * idx;
    k_wpart<<<256, 256, 0, stream>>>(w, part);
    k_wfinal<<<1, 256, 0, stream>>>(part, ab);
    k_binarize<<<4096, 256, 0, stream>>>(w, ab, wb + (size_t)idx * 1048576);
  }

  hipMemcpyAsync(xA, x0, 8388608, hipMemcpyDeviceToDevice, stream);

  for (int L = 0; L < 6; ++L) {
    k_pack<<<8192, 256, 0, stream>>>(xA, xh, xht);
    k_flash<<<dim3(8, 32), 256, 0, stream>>>(xh, xht, xA, xB);            // xB = attend(x)+x
    k_ln<<<2048, 256, 0, stream>>>(xB, lng + L * 1024, lnb + L * 1024, hbuf, rowam);
    k_rmax<<<1, 256, 0, stream>>>(rowam, 2048, gam);                      // gamma1
    k_quant<<<8192, 256, 0, stream>>>(hbuf, gam, hq);
    k_gemm<<<dim3(16, 8), 256, 0, stream>>>(hq, wb + (size_t)(2 * L) * 1048576,
                                            scal + 2 * (2 * L) + 1, gam,
                                            nullptr, hbuf, part, 1);      // gelu + absmax partials
    k_rmax<<<1, 256, 0, stream>>>(part, 128, gam);                        // gamma2
    k_quant<<<8192, 256, 0, stream>>>(hbuf, gam, hq);
    k_gemm<<<dim3(16, 8), 256, 0, stream>>>(hq, wb + (size_t)(2 * L + 1) * 1048576,
                                            scal + 2 * (2 * L + 1) + 1, gam,
                                            xB, xA, nullptr, 0);          // +residual -> xA
  }
  k_ln<<<2048, 256, 0, stream>>>(xA, fg, fb, (float*)d_out, nullptr);
}
